// AttentionBlock_33097017983007
// MI455X (gfx1250) — matrix operand order repaired and checked by reading, not compiled, not run on hardware
//
#include <hip/hip_runtime.h>

typedef __attribute__((ext_vector_type(16))) __bf16 v16bf;
typedef __attribute__((ext_vector_type(8)))  __bf16 v8bf;
typedef __attribute__((ext_vector_type(8)))  float  v8f;

static constexpr int Bn = 8, Ln = 2048, Cn = 512, Gn = 8, CG = Cn / Gn;
static constexpr int NK = 128;                 // kv tile for flash
static constexpr float EPSV = 1e-3f;

union V16U { v16bf v; v8bf h[2]; };

__device__ __forceinline__ unsigned short f32_to_bf16(float f) {
  unsigned int u = __float_as_uint(f);
  u += 0x7FFFu + ((u >> 16) & 1u);            // round-to-nearest-even
  return (unsigned short)(u >> 16);
}

// A operand (16xK tile of row-major [M,K] memory): lane<16 holds row, K {0..7,16..23};
// lane>=16 holds K {8..15,24..31}.
__device__ __forceinline__ v16bf frag_a(const unsigned short* base, int ld, int k0, int lane) {
  const unsigned short* p = base + (size_t)(lane & 15) * ld + k0 + ((lane & 16) ? 8 : 0);
  V16U u;
  u.h[0] = *(const v8bf*)(p);
  u.h[1] = *(const v8bf*)(p + 16);
  return u.v;
}

// B operand from [N,K] row-major memory (NT gemm): lane<16 holds N=lane K {0..7,16..23},
// lane>=16 holds K {8..15,24..31}.
__device__ __forceinline__ v16bf frag_b(const unsigned short* base, int ld, int k0, int lane) {
  const unsigned short* p = base + (size_t)(lane & 15) * ld + k0 + ((lane & 16) ? 8 : 0);
  V16U u;
  u.h[0] = *(const v8bf*)(p);
  u.h[1] = *(const v8bf*)(p + 16);
  return u.v;
}

#define WMMA_BF16(A, B, CC) \
  __builtin_amdgcn_wmma_f32_16x16x32_bf16(false, (A), false, (B), (short)0, (CC), false, false)

// ---------------- GroupNorm ----------------
__global__ void gn_stats(const float* __restrict__ x, float* __restrict__ mean,
                         float* __restrict__ rstd) {
  int bg = blockIdx.x;                 // B*G = 64
  int b = bg / Gn, g = bg % Gn;
  const float* xp = x + (size_t)b * Ln * Cn + g * CG;
  float s = 0.f, ss = 0.f;
  for (int i = threadIdx.x; i < Ln * CG; i += blockDim.x) {
    int l = i >> 6, c = i & (CG - 1);
    float v = xp[(size_t)l * Cn + c];
    s += v; ss += v * v;
  }
  __shared__ float shs[256], shss[256];
  shs[threadIdx.x] = s; shss[threadIdx.x] = ss;
  __syncthreads();
  for (int off = 128; off > 0; off >>= 1) {
    if ((int)threadIdx.x < off) {
      shs[threadIdx.x] += shs[threadIdx.x + off];
      shss[threadIdx.x] += shss[threadIdx.x + off];
    }
    __syncthreads();
  }
  if (threadIdx.x == 0) {
    float inv_n = 1.f / (float)(Ln * CG);
    float m = shs[0] * inv_n;
    float var = shss[0] * inv_n - m * m;
    mean[bg] = m;
    rstd[bg] = rsqrtf(var + EPSV);
  }
}

__global__ void gn_apply(const float* __restrict__ x, const float* __restrict__ gamma,
                         const float* __restrict__ beta, const float* __restrict__ mean,
                         const float* __restrict__ rstd, float* __restrict__ hf,
                         unsigned short* __restrict__ hb) {
  size_t i = (size_t)blockIdx.x * blockDim.x + threadIdx.x;  // B*L*C elements
  size_t c = i & (Cn - 1);
  size_t bl = i >> 9;           // /C
  size_t b = bl >> 11;          // /L
  int bg = (int)(b * Gn + (c >> 6));
  float v = (x[i] - mean[bg]) * rstd[bg] * gamma[c] + beta[c];
  hf[i] = v;
  hb[i] = f32_to_bf16(v);
}

// weights [K,N] f32 -> [N,K] bf16
__global__ void cast_w(const float* __restrict__ w, unsigned short* __restrict__ wT) {
  int i = blockIdx.x * blockDim.x + threadIdx.x;  // C*C
  int k = i >> 9, n = i & (Cn - 1);
  wT[(size_t)n * Cn + k] = f32_to_bf16(w[i]);
}

// ---------------- QKV GEMM: [L,C] x [C,C] ----------------
__global__ void __launch_bounds__(256)
qkv_gemm(const unsigned short* __restrict__ h, const unsigned short* __restrict__ wT,
         const float* __restrict__ bias, unsigned short* __restrict__ out,
         int transpose_out) {
  int b = blockIdx.y;
  int wave = threadIdx.x >> 5, lane = threadIdx.x & 31;
  int tile = blockIdx.x * 8 + wave;     // 64 x 16 tiles of 32x32
  int m0 = (tile >> 4) * 32, n0 = (tile & 15) * 32;
  const unsigned short* A = h + (size_t)b * Ln * Cn;
  v8f acc[2][2] = {};
  for (int k0 = 0; k0 < Cn; k0 += 32) {
    v16bf a0 = frag_a(A + (size_t)m0 * Cn, Cn, k0, lane);
    v16bf a1 = frag_a(A + (size_t)(m0 + 16) * Cn, Cn, k0, lane);
    v16bf b0 = frag_b(wT + (size_t)n0 * Cn, Cn, k0, lane);
    v16bf b1 = frag_b(wT + (size_t)(n0 + 16) * Cn, Cn, k0, lane);
    acc[0][0] = WMMA_BF16(a0, b0, acc[0][0]);
    acc[0][1] = WMMA_BF16(a0, b1, acc[0][1]);
    acc[1][0] = WMMA_BF16(a1, b0, acc[1][0]);
    acc[1][1] = WMMA_BF16(a1, b1, acc[1][1]);
  }
  int halfsel = (lane >> 4) & 1, lrow = lane & 15;
  for (int mi = 0; mi < 2; ++mi)
    for (int ni = 0; ni < 2; ++ni)
#pragma unroll
      for (int i = 0; i < 8; ++i) {
        int row = m0 + mi * 16 + i + halfsel * 8;
        int col = n0 + ni * 16 + lrow;
        unsigned short v = f32_to_bf16(acc[mi][ni][i] + bias[col]);
        if (!transpose_out)
          out[(size_t)b * Ln * Cn + (size_t)row * Cn + col] = v;
        else
          out[(size_t)b * Cn * Ln + (size_t)col * Ln + row] = v;  // v^T [C,L]
      }
}

// ---------------- Fused flash attention + output projection + residual ----------------
__global__ void __launch_bounds__(256)
flash_attn_proj(const unsigned short* __restrict__ q, const unsigned short* __restrict__ k,
                const unsigned short* __restrict__ vT, const unsigned short* __restrict__ wpT,
                const float* __restrict__ bp, const float* __restrict__ hf,
                float* __restrict__ out) {
  int b = blockIdx.y;
  int q0 = blockIdx.x * 16;
  int wave = threadIdx.x >> 5, lane = threadIdx.x & 31;
  int halfsel = (lane >> 4) & 1, lrow = lane & 15;

  __shared__ __align__(16) unsigned short Qs[16 * Cn];  // Q tile bf16; reused for O bf16
  __shared__ __align__(16) float S[16 * NK];            // score tile
  __shared__ __align__(16) unsigned short P[16 * NK];   // probs bf16
  __shared__ float red[16 * 16];                        // partial reduce [seg][row]
  __shared__ float mrow[16], lsum[16], arow[16];

  // stage Q tile into LDS (coalesced dword copies)
  {
    const unsigned int* src = (const unsigned int*)(q + (size_t)b * Ln * Cn + (size_t)q0 * Cn);
    unsigned int* dst = (unsigned int*)Qs;
    for (int i = threadIdx.x; i < 16 * Cn / 2; i += 256) dst[i] = src[i];
  }
  if (threadIdx.x < 16) { mrow[threadIdx.x] = -3.0e38f; lsum[threadIdx.x] = 0.f; }
  __syncthreads();

  v8f acc[4] = {};
  const unsigned short* kb = k + (size_t)b * Ln * Cn;
  const unsigned short* vb = vT + (size_t)b * Cn * Ln;
  const float scale = 0.044194173824159216f;  // 512^-0.5

  for (int t = 0; t < Ln / NK; ++t) {
    int kv0 = t * NK;
    // prefetch next K tile rows into L2
    if (t + 1 < Ln / NK)
      __builtin_prefetch(kb + (size_t)(kv0 + NK + 16 * wave + lrow) * Cn, 0, 1);

    // S = Q K^T : wave computes score columns [16*wave, 16*wave+16)
    v8f sacc = {};
    const unsigned short* krows = kb + (size_t)(kv0 + 16 * wave) * Cn;
    for (int k0 = 0; k0 < Cn; k0 += 32) {
      v16bf aq = frag_a(Qs, Cn, k0, lane);
      v16bf bk = frag_b(krows, Cn, k0, lane);
      sacc = WMMA_BF16(aq, bk, sacc);
    }
#pragma unroll
    for (int i = 0; i < 8; ++i)
      S[(i + halfsel * 8) * NK + 16 * wave + lrow] = sacc[i] * scale;
    __syncthreads();

    // online softmax: 16 threads/row, 8 cols each
    {
      int r = threadIdx.x & 15, sg = threadIdx.x >> 4;
      float mx = -3.0e38f;
      for (int j2 = sg * 8; j2 < sg * 8 + 8; ++j2) mx = fmaxf(mx, S[r * NK + j2]);
      red[sg * 16 + r] = mx;
    }
    __syncthreads();
    if (threadIdx.x < 16) {
      int r = threadIdx.x;
      float mx = mrow[r];
      for (int sg = 0; sg < 16; ++sg) mx = fmaxf(mx, red[sg * 16 + r]);
      arow[r] = __expf(mrow[r] - mx);
      mrow[r] = mx;
    }
    __syncthreads();
    {
      int r = threadIdx.x & 15, sg = threadIdx.x >> 4;
      float mx = mrow[r], s = 0.f;
      for (int j2 = sg * 8; j2 < sg * 8 + 8; ++j2) {
        float p = __expf(S[r * NK + j2] - mx);
        s += p;
        P[r * NK + j2] = f32_to_bf16(p);
      }
      red[sg * 16 + r] = s;
    }
    __syncthreads();
    if (threadIdx.x < 16) {
      int r = threadIdx.x;
      float s = 0.f;
      for (int sg = 0; sg < 16; ++sg) s += red[sg * 16 + r];
      lsum[r] = lsum[r] * arow[r] + s;
    }
    __syncthreads();

    // rescale O and accumulate P·V (wave owns O cols [64*wave, 64*wave+64))
#pragma unroll
    for (int j = 0; j < 4; ++j)
#pragma unroll
      for (int i = 0; i < 8; ++i) acc[j][i] *= arow[i + halfsel * 8];
#pragma unroll
    for (int kj = 0; kj < 4; ++kj) {
      v16bf ap = frag_a(P, NK, kj * 32, lane);
#pragma unroll
      for (int j = 0; j < 4; ++j) {
        const unsigned short* vrows = vb + (size_t)(64 * wave + 16 * j) * Ln + kv0;
        v16bf bv = frag_b(vrows, Ln, kj * 32, lane);
        acc[j] = WMMA_BF16(ap, bv, acc[j]);
      }
    }
    __syncthreads();
  }

  // normalize O by row sums, stage bf16 into Qs (Q no longer needed)
#pragma unroll
  for (int j = 0; j < 4; ++j)
#pragma unroll
    for (int i = 0; i < 8; ++i) {
      int row = i + halfsel * 8;
      Qs[row * Cn + 64 * wave + 16 * j + lrow] = f32_to_bf16(acc[j][i] / lsum[row]);
    }
  __syncthreads();

  // fused projection: out = h + O @ wp + bp ; wave owns cols [64*wave, +64)
  v8f pacc[4] = {};
  for (int k0 = 0; k0 < Cn; k0 += 32) {
    v16bf ao = frag_a(Qs, Cn, k0, lane);
#pragma unroll
    for (int j = 0; j < 4; ++j) {
      v16bf bw = frag_b(wpT + (size_t)(64 * wave + 16 * j) * Cn, Cn, k0, lane);
      pacc[j] = WMMA_BF16(ao, bw, pacc[j]);
    }
  }
  const float* hrow = hf + (size_t)b * Ln * Cn + (size_t)q0 * Cn;
  float* orow = out + (size_t)b * Ln * Cn + (size_t)q0 * Cn;
#pragma unroll
  for (int j = 0; j < 4; ++j)
#pragma unroll
    for (int i = 0; i < 8; ++i) {
      int row = i + halfsel * 8;
      int col = 64 * wave + 16 * j + lrow;
      orow[(size_t)row * Cn + col] = hrow[(size_t)row * Cn + col] + pacc[j][i] + bp[col];
    }
}

extern "C" void kernel_launch(void* const* d_in, const int* in_sizes, int n_in,
                              void* d_out, int out_size, void* d_ws, size_t ws_size,
                              hipStream_t stream) {
  (void)in_sizes; (void)n_in; (void)out_size; (void)ws_size;
  const float* x     = (const float*)d_in[0];
  const float* gamma = (const float*)d_in[1];
  const float* beta  = (const float*)d_in[2];
  const float* wq    = (const float*)d_in[3];
  const float* bq    = (const float*)d_in[4];
  const float* wk    = (const float*)d_in[5];
  const float* bk    = (const float*)d_in[6];
  const float* wv    = (const float*)d_in[7];
  const float* bv    = (const float*)d_in[8];
  const float* wp    = (const float*)d_in[9];
  const float* bp    = (const float*)d_in[10];
  float* out = (float*)d_out;

  char* ws = (char*)d_ws;
  size_t off = 0;
  auto alloc = [&](size_t bytes) -> void* {
    off = (off + 255) & ~(size_t)255;
    void* p = ws + off;
    off += bytes;
    return p;
  };
  const size_t NE = (size_t)Bn * Ln * Cn;
  float* mean = (float*)alloc(Bn * Gn * sizeof(float));
  float* rstd = (float*)alloc(Bn * Gn * sizeof(float));
  float* hf = (float*)alloc(NE * sizeof(float));
  unsigned short* hb  = (unsigned short*)alloc(NE * 2);
  unsigned short* wqT = (unsigned short*)alloc((size_t)Cn * Cn * 2);
  unsigned short* wkT = (unsigned short*)alloc((size_t)Cn * Cn * 2);
  unsigned short* wvT = (unsigned short*)alloc((size_t)Cn * Cn * 2);
  unsigned short* wpT = (unsigned short*)alloc((size_t)Cn * Cn * 2);
  unsigned short* qb  = (unsigned short*)alloc(NE * 2);
  unsigned short* kbf = (unsigned short*)alloc(NE * 2);
  unsigned short* vTb = (unsigned short*)alloc(NE * 2);

  gn_stats<<<Bn * Gn, 256, 0, stream>>>(x, mean, rstd);
  gn_apply<<<(int)(NE / 256), 256, 0, stream>>>(x, gamma, beta, mean, rstd, hf, hb);
  cast_w<<<Cn * Cn / 256, 256, 0, stream>>>(wq, wqT);
  cast_w<<<Cn * Cn / 256, 256, 0, stream>>>(wk, wkT);
  cast_w<<<Cn * Cn / 256, 256, 0, stream>>>(wv, wvT);
  cast_w<<<Cn * Cn / 256, 256, 0, stream>>>(wp, wpT);
  qkv_gemm<<<dim3(128, Bn), 256, 0, stream>>>(hb, wqT, bq, qb, 0);
  qkv_gemm<<<dim3(128, Bn), 256, 0, stream>>>(hb, wkT, bk, kbf, 0);
  qkv_gemm<<<dim3(128, Bn), 256, 0, stream>>>(hb, wvT, bv, vTb, 1);
  flash_attn_proj<<<dim3(Ln / 16, Bn), 256, 0, stream>>>(qb, kbf, vTb, wpT, bp, hf, out);
}
